// Seq2Seq_1872605741492
// MI455X (gfx1250) — matrix operand order repaired and checked by reading, not compiled, not run on hardware
//
#include <hip/hip_runtime.h>
#include <hip/hip_bf16.h>
#include <math.h>
#include <stdint.h>

// ---------------------------------------------------------------------------
// Seq2Seq video captioner for MI455X (gfx1250, wave32, WMMA).
// - All GEMMs on v_wmma_f32_16x16x32_bf16, fp32 accumulate.
// - Weights converted to bf16 once per launch; W2 (16 MB bf16) stays L2-resident
//   across the 49 decode steps (192 MB L2).
// - Shared activation panels ([64x256] x / h matrices read by every wave of a
//   block) are broadcast through LDS with GLOBAL_LOAD_ASYNC_TO_LDS_B128 and
//   s_wait_asynccnt, removing 8x duplicated VMEM traffic.
// ---------------------------------------------------------------------------

typedef __bf16 bf16_t;
typedef __attribute__((ext_vector_type(16))) __bf16 v16bf;
typedef __attribute__((ext_vector_type(8)))  __bf16 v8bf;
typedef __attribute__((ext_vector_type(8)))  float  v8f;

#define B_   64
#define H_   256
#define VS_  80
#define CS_  50
#define V_   32000
#define FD_  4096
#define T1_  (VS_ + CS_ - 1)   /* 129 timesteps for LSTM1 */
#define DEC_ (CS_ - 1)         /* 49 decode steps */

__device__ __forceinline__ v8f wmma_bf16(v16bf a, v16bf b, v8f c) {
  // 8-arg form: (neg_a, A, neg_b, B, c_mod, C, reuse_a, reuse_b)
  return __builtin_amdgcn_wmma_f32_16x16x32_bf16(false, a, false, b,
                                                 (short)0, c, false, false);
}

// Async copy of one 16B chunk from global to LDS (tracked by ASYNCcnt).
// lds_off = wave-relative LDS byte address (low 32 bits of a __shared__ ptr).
__device__ __forceinline__ void async_ld_b128(uint32_t lds_off, const void* gaddr) {
  asm volatile("global_load_async_to_lds_b128 %0, %1, off"
               :: "v"(lds_off), "v"(gaddr) : "memory");
}
__device__ __forceinline__ void wait_asynccnt0() {
  asm volatile("s_wait_asynccnt 0x0" ::: "memory");
}

// ---- fragment loaders -----------------------------------------------------
// A (16x32 MxK), 16-bit layout: lane group g=lane>>4;
//   element j<8  -> K = k0 + 8g + j ; element j>=8 -> K = k0 + 16 + 8g + (j-8)
__device__ __forceinline__ v16bf ld_a_bf16(const bf16_t* row, int k0, int g) {
  v8bf lo = *(const v8bf*)(row + k0 + 8 * g);
  v8bf hi = *(const v8bf*)(row + k0 + 16 + 8 * g);
  v16bf r;
#pragma unroll
  for (int j = 0; j < 8; ++j) { r[j] = lo[j]; r[j + 8] = hi[j]; }
  return r;
}

__device__ __forceinline__ v16bf ld_a_f32(const float* row, int k0, int g) {
  v8f lo = *(const v8f*)(row + k0 + 8 * g);
  v8f hi = *(const v8f*)(row + k0 + 16 + 8 * g);
  v16bf r;
#pragma unroll
  for (int j = 0; j < 8; ++j) { r[j] = (bf16_t)lo[j]; r[j + 8] = (bf16_t)hi[j]; }
  return r;
}

// B (32x16 KxN) sourced from row-major W[N][K] (out = A @ W^T): per-lane row
// is N = lane&15; element j<8 -> K = k0 + 8g + j ; j>=8 -> K = k0 + 16 + 8g + (j-8).
__device__ __forceinline__ v16bf ld_b_bf16(const bf16_t* row, int k0, int g) {
  v8bf lo = *(const v8bf*)(row + k0 + 8 * g);
  v8bf hi = *(const v8bf*)(row + k0 + 16 + 8 * g);
  v16bf r;
#pragma unroll
  for (int j = 0; j < 8; ++j) { r[j] = lo[j]; r[j + 8] = hi[j]; }
  return r;
}

__device__ __forceinline__ float sigmoidf_(float x) {
  return 1.0f / (1.0f + __expf(-x));
}

// ---- weight conversion ----------------------------------------------------
__global__ void __launch_bounds__(256)
cvt_f32_bf16_kernel(const float* __restrict__ src, bf16_t* __restrict__ dst, int n) {
  int i = blockIdx.x * 256 + threadIdx.x;
  if (i < n) dst[i] = (bf16_t)src[i];
}

// ---- projection: v = video @ W1^T + b1, stored time-major bf16 ------------
// grid: 160 blocks x 256 thr = 1280 waves; wave -> (mg 0..79, tn 0..15);
// wave computes 4 M-tiles (64 rows) x one 16-col tile, K=4096.
__global__ void __launch_bounds__(256)
proj_gemm_kernel(const float* __restrict__ video, const bf16_t* __restrict__ W1bf,
                 const float* __restrict__ b1, bf16_t* __restrict__ vbuf) {
  const int wid  = blockIdx.x * 8 + (threadIdx.x >> 5);
  const int tn   = wid & 15;
  const int mg   = wid >> 4;          // 0..79: group of 64 rows
  const int lane = threadIdx.x & 31;
  const int g    = lane >> 4, ln = lane & 15;
  const int n0   = tn * 16;

  v8f acc[4] = {v8f{}, v8f{}, v8f{}, v8f{}};
  const bf16_t* wrow = W1bf + (size_t)(n0 + ln) * FD_;
  for (int k = 0; k < FD_; k += 32) {
    __builtin_prefetch(wrow + k + 64, 0, 0);   // global_prefetch_b8
    v16bf bf = ld_b_bf16(wrow, k, g);
#pragma unroll
    for (int mt = 0; mt < 4; ++mt) {
      const int m = mg * 64 + mt * 16 + ln;
      v16bf af = ld_a_f32(video + (size_t)m * FD_, k, g);
      acc[mt] = wmma_bf16(af, bf, acc[mt]);
    }
  }
  const float bias = b1[n0 + ln];
#pragma unroll
  for (int mt = 0; mt < 4; ++mt)
#pragma unroll
    for (int r = 0; r < 8; ++r) {
      const int m  = mg * 64 + mt * 16 + r + 8 * g;   // row of flattened video
      const int tt = m % VS_;                          // timestep
      const int bb = m / VS_;                          // batch
      vbuf[((size_t)tt * B_ + bb) * H_ + n0 + ln] = (bf16_t)(acc[mt][r] + bias);
    }
}

// ---- LSTM gate GEMM: gates[64,1024] = sum_i Ai @ Wi^T + H @ Wh^T ----------
// grid: 8 blocks x 256 thr = 64 waves; wave owns one 16-col tile of the 1024
// gate columns and all 4 M-tiles. A0 and H ([64x256] bf16, 32 KB each) are
// broadcast to the block through LDS via async global->LDS copies.
__global__ void __launch_bounds__(256)
gate_gemm_kernel(float* __restrict__ gates,
                 const bf16_t* __restrict__ A0, const bf16_t* __restrict__ W0,
                 int ldw0, int K0,
                 const bf16_t* __restrict__ A1, const bf16_t* __restrict__ W1w,
                 int ldw1, int K1,
                 const bf16_t* __restrict__ Hb, const bf16_t* __restrict__ Wh) {
  __shared__ bf16_t sA0[B_ * H_];   // 32 KB
  __shared__ bf16_t sH [B_ * H_];   // 32 KB

  const int tid  = threadIdx.x;
  const int lane = tid & 31;
  const int g    = lane >> 4, ln = lane & 15;
  const int tn   = blockIdx.x * 8 + (tid >> 5);  // 0..63
  const int n0   = tn * 16;

  // stage the shared activation panels: 2048 x 16B chunks each, 8 per thread
#pragma unroll
  for (int j = 0; j < 8; ++j) {
    const int chunk = tid + j * 256;             // 0..2047
    async_ld_b128((uint32_t)(uintptr_t)&sA0[chunk * 8], A0 + chunk * 8);
    async_ld_b128((uint32_t)(uintptr_t)&sH [chunk * 8], Hb + chunk * 8);
  }
  wait_asynccnt0();
  __syncthreads();

  v8f acc[4] = {v8f{}, v8f{}, v8f{}, v8f{}};

  {  // input stream 0 (from LDS)
    const bf16_t* wrow = W0 + (size_t)(n0 + ln) * ldw0;
    for (int k = 0; k < K0; k += 32) {
      __builtin_prefetch(wrow + k + 32, 0, 0);
      v16bf bf = ld_b_bf16(wrow, k, g);
#pragma unroll
      for (int mt = 0; mt < 4; ++mt) {
        v16bf af = ld_a_bf16(&sA0[(mt * 16 + ln) * H_], k, g);
        acc[mt] = wmma_bf16(af, bf, acc[mt]);
      }
    }
  }
  if (A1) {  // optional input stream 1 (decoder context) -- uniform branch
    const bf16_t* wrow = W1w + (size_t)(n0 + ln) * ldw1;
    for (int k = 0; k < K1; k += 32) {
      v16bf bf = ld_b_bf16(wrow, k, g);
#pragma unroll
      for (int mt = 0; mt < 4; ++mt) {
        v16bf af = ld_a_bf16(A1 + (size_t)(mt * 16 + ln) * H_, k, g);
        acc[mt] = wmma_bf16(af, bf, acc[mt]);
      }
    }
  }
  {  // recurrent stream (from LDS)
    const bf16_t* wrow = Wh + (size_t)(n0 + ln) * H_;
    for (int k = 0; k < H_; k += 32) {
      v16bf bf = ld_b_bf16(wrow, k, g);
#pragma unroll
      for (int mt = 0; mt < 4; ++mt) {
        v16bf af = ld_a_bf16(&sH[(mt * 16 + ln) * H_], k, g);
        acc[mt] = wmma_bf16(af, bf, acc[mt]);
      }
    }
  }
#pragma unroll
  for (int mt = 0; mt < 4; ++mt)
#pragma unroll
    for (int r = 0; r < 8; ++r) {
      const int m = mt * 16 + r + 8 * g;
      gates[(size_t)m * (4 * H_) + n0 + ln] = acc[mt][r];
    }
}

// ---- LSTM cell nonlinearity (PyTorch gate order i,f,g,o) ------------------
__global__ void __launch_bounds__(256)
lstm_cell_kernel(const float* __restrict__ gates, const float* __restrict__ bias,
                 float* __restrict__ c, bf16_t* __restrict__ h,
                 bf16_t* __restrict__ hcopy) {
  const int idx = blockIdx.x * 256 + threadIdx.x;   // 64*256 = 16384 threads
  const int b = idx >> 8, n = idx & 255;
  const float* gr = gates + (size_t)b * (4 * H_);
  const float gi = gr[n]           + bias[n];
  const float gf = gr[H_ + n]      + bias[H_ + n];
  const float gg = gr[2 * H_ + n]  + bias[2 * H_ + n];
  const float go = gr[3 * H_ + n]  + bias[3 * H_ + n];
  const float cn = sigmoidf_(gf) * c[idx] + sigmoidf_(gi) * tanhf(gg);
  const float hn = sigmoidf_(go) * tanhf(cn);
  c[idx] = cn;
  const bf16_t hb = (bf16_t)hn;
  h[idx] = hb;
  if (hcopy) hcopy[idx] = hb;
}

// ---- embedding gather: x = E[tok] (f32 -> bf16 on the fly) ----------------
__global__ void __launch_bounds__(256)
embed_kernel(const float* __restrict__ E, const int* __restrict__ tok,
             bf16_t* __restrict__ x) {
  const int idx = blockIdx.x * 256 + threadIdx.x;   // 16384 threads
  const int b = idx >> 8, n = idx & 255;
  x[idx] = (bf16_t)E[(size_t)tok[b] * H_ + n];
}

// ---- logits: out = H @ W2^T + b2, written straight into cap_prob ----------
// grid: 250 blocks x 256 thr = 2000 waves; wave owns one 16-col tile of the
// 32000 vocab columns and all 4 M-tiles. K = 256. H is broadcast via LDS so
// the full VMEM path is dedicated to streaming W2 from L2.
__global__ void __launch_bounds__(256)
logits_gemm_kernel(const bf16_t* __restrict__ Hb, const bf16_t* __restrict__ W2bf,
                   const float* __restrict__ b2, float* __restrict__ out,
                   long long row_stride) {
  __shared__ bf16_t sH[B_ * H_];   // 32 KB

  const int tid  = threadIdx.x;
  const int wid  = blockIdx.x * 8 + (tid >> 5);  // 0..1999 (N tile)
  const int lane = tid & 31;
  const int g    = lane >> 4, ln = lane & 15;
  const int n0   = wid * 16;

#pragma unroll
  for (int j = 0; j < 8; ++j) {
    const int chunk = tid + j * 256;             // 0..2047
    async_ld_b128((uint32_t)(uintptr_t)&sH[chunk * 8], Hb + chunk * 8);
  }
  wait_asynccnt0();
  __syncthreads();

  v8f acc[4] = {v8f{}, v8f{}, v8f{}, v8f{}};
  const bf16_t* wrow = W2bf + (size_t)(n0 + ln) * H_;
  for (int k = 0; k < H_; k += 32) {
    __builtin_prefetch(wrow + k + 32, 0, 0);
    v16bf bf = ld_b_bf16(wrow, k, g);
#pragma unroll
    for (int mt = 0; mt < 4; ++mt) {
      v16bf af = ld_a_bf16(&sH[(mt * 16 + ln) * H_], k, g);
      acc[mt] = wmma_bf16(af, bf, acc[mt]);
    }
  }
  const float bias = b2[n0 + ln];
#pragma unroll
  for (int mt = 0; mt < 4; ++mt)
#pragma unroll
    for (int r = 0; r < 8; ++r) {
      const int m = mt * 16 + r + 8 * g;     // batch row
      out[(size_t)m * row_stride + n0 + ln] = acc[mt][r] + bias;
    }
}

// ---- greedy argmax over vocab (ties -> lowest index, matching jnp.argmax) -
__global__ void __launch_bounds__(256)
argmax_kernel(const float* __restrict__ logits, long long row_stride,
              int* __restrict__ tok, float* __restrict__ caption, int t) {
  const int b = blockIdx.x;
  const float* row = logits + (size_t)b * row_stride;
  float best = -INFINITY;
  int bi = 0x7fffffff;
  for (int v = threadIdx.x; v < V_; v += 256) {
    const float x = row[v];
    if (x > best) { best = x; bi = v; }
  }
  __shared__ float sv[256];
  __shared__ int   si[256];
  sv[threadIdx.x] = best; si[threadIdx.x] = bi;
  __syncthreads();
  for (int s = 128; s > 0; s >>= 1) {
    if (threadIdx.x < s) {
      const float ov = sv[threadIdx.x + s];
      const int   oi = si[threadIdx.x + s];
      if (ov > sv[threadIdx.x] ||
          (ov == sv[threadIdx.x] && oi < si[threadIdx.x])) {
        sv[threadIdx.x] = ov; si[threadIdx.x] = oi;
      }
    }
    __syncthreads();
  }
  if (threadIdx.x == 0) {
    tok[b] = si[0];
    caption[(size_t)b * DEC_ + t] = (float)si[0];
  }
}

// ---------------------------------------------------------------------------
extern "C" void kernel_launch(void* const* d_in, const int* in_sizes, int n_in,
                              void* d_out, int out_size, void* d_ws, size_t ws_size,
                              hipStream_t stream) {
  const float* video = (const float*)d_in[0];
  const float* W1    = (const float*)d_in[1];
  const float* b1    = (const float*)d_in[2];
  const float* Wih1  = (const float*)d_in[3];
  const float* Whh1  = (const float*)d_in[4];
  const float* b_l1  = (const float*)d_in[5];
  const float* Wih2  = (const float*)d_in[6];
  const float* Whh2  = (const float*)d_in[7];
  const float* b_l2  = (const float*)d_in[8];
  const float* E     = (const float*)d_in[9];
  const float* W2    = (const float*)d_in[10];
  const float* b2    = (const float*)d_in[11];
  float* out = (float*)d_out;

  // workspace carve-out (256B aligned)
  char* ws = (char*)d_ws;
  size_t off = 0;
  auto carve = [&](size_t bytes) {
    void* p = ws + off;
    off = (off + bytes + 255) & ~(size_t)255;
    return p;
  };
  bf16_t* W1bf   = (bf16_t*)carve((size_t)H_ * FD_ * 2);
  bf16_t* Wih1bf = (bf16_t*)carve((size_t)4 * H_ * H_ * 2);
  bf16_t* Whh1bf = (bf16_t*)carve((size_t)4 * H_ * H_ * 2);
  bf16_t* Wih2bf = (bf16_t*)carve((size_t)4 * H_ * 2 * H_ * 2);
  bf16_t* Whh2bf = (bf16_t*)carve((size_t)4 * H_ * H_ * 2);
  bf16_t* W2bf   = (bf16_t*)carve((size_t)V_ * H_ * 2);
  bf16_t* vbuf   = (bf16_t*)carve((size_t)T1_ * B_ * H_ * 2);  // time-major proj
  bf16_t* vidout = (bf16_t*)carve((size_t)T1_ * B_ * H_ * 2);  // LSTM1 hiddens
  float*  gates  = (float*) carve((size_t)B_ * 4 * H_ * 4);
  bf16_t* h1bf   = (bf16_t*)carve((size_t)B_ * H_ * 2);
  float*  c1     = (float*) carve((size_t)B_ * H_ * 4);
  bf16_t* h2bf   = (bf16_t*)carve((size_t)B_ * H_ * 2);
  float*  c2     = (float*) carve((size_t)B_ * H_ * 4);
  bf16_t* xemb   = (bf16_t*)carve((size_t)B_ * H_ * 2);
  int*    tok    = (int*)   carve((size_t)B_ * 4);

  auto cvt = [&](const float* s, bf16_t* d, int n) {
    cvt_f32_bf16_kernel<<<(n + 255) / 256, 256, 0, stream>>>(s, d, n);
  };
  cvt(W1,   W1bf,   H_ * FD_);
  cvt(Wih1, Wih1bf, 4 * H_ * H_);
  cvt(Whh1, Whh1bf, 4 * H_ * H_);
  cvt(Wih2, Wih2bf, 4 * H_ * 2 * H_);
  cvt(Whh2, Whh2bf, 4 * H_ * H_);
  cvt(W2,   W2bf,   V_ * H_);

  // zero recurrent state, token seed (BOS=0), and padded video steps
  hipMemsetAsync(h1bf, 0, (size_t)B_ * H_ * 2, stream);
  hipMemsetAsync(c1,   0, (size_t)B_ * H_ * 4, stream);
  hipMemsetAsync(h2bf, 0, (size_t)B_ * H_ * 2, stream);
  hipMemsetAsync(c2,   0, (size_t)B_ * H_ * 4, stream);
  hipMemsetAsync(tok,  0, (size_t)B_ * 4, stream);
  hipMemsetAsync(vbuf + (size_t)VS_ * B_ * H_, 0,
                 (size_t)(T1_ - VS_) * B_ * H_ * 2, stream);

  // 1) projection GEMM
  proj_gemm_kernel<<<160, 256, 0, stream>>>(video, W1bf, b1, vbuf);

  // 2) LSTM1 over 129 steps (h written into vidout[t])
  for (int t = 0; t < T1_; ++t) {
    gate_gemm_kernel<<<8, 256, 0, stream>>>(
        gates, vbuf + (size_t)t * B_ * H_, Wih1bf, H_, H_,
        nullptr, nullptr, 0, 0, h1bf, Whh1bf);
    lstm_cell_kernel<<<64, 256, 0, stream>>>(
        gates, b_l1, c1, h1bf, vidout + (size_t)t * B_ * H_);
  }

  // 3) LSTM2 encoder over 80 steps: x = [zeros | vidout[t]] -> only the
  //    second half of Wih2's columns contributes (offset 256, row stride 512).
  for (int t = 0; t < VS_; ++t) {
    gate_gemm_kernel<<<8, 256, 0, stream>>>(
        gates, vidout + (size_t)t * B_ * H_, Wih2bf + H_, 2 * H_, H_,
        nullptr, nullptr, 0, 0, h2bf, Whh2bf);
    lstm_cell_kernel<<<64, 256, 0, stream>>>(gates, b_l2, c2, h2bf, nullptr);
  }

  // 4) greedy decode, 49 steps
  float* cap_prob = out + (size_t)B_ * DEC_;   // caption first, then probs
  for (int t = 0; t < DEC_; ++t) {
    embed_kernel<<<64, 256, 0, stream>>>(E, tok, xemb);
    gate_gemm_kernel<<<8, 256, 0, stream>>>(
        gates, xemb, Wih2bf, 2 * H_, H_,
        vidout + (size_t)(VS_ + t) * B_ * H_, Wih2bf + H_, 2 * H_, H_,
        h2bf, Whh2bf);
    lstm_cell_kernel<<<64, 256, 0, stream>>>(gates, b_l2, c2, h2bf, nullptr);
    float* prob_t = cap_prob + (size_t)t * V_;   // [b][t][v] -> b*49*V + t*V + v
    logits_gemm_kernel<<<250, 256, 0, stream>>>(
        h2bf, W2bf, b2, prob_t, (long long)DEC_ * V_);
    argmax_kernel<<<64, 256, 0, stream>>>(
        prob_t, (long long)DEC_ * V_, tok, out, t);
  }
}
